// SegmentVQVAE_60413009985829
// MI455X (gfx1250) — matrix operand order repaired and checked by reading, not compiled, not run on hardware
//
#include <hip/hip_runtime.h>
#include <math.h>

// ---------------------------------------------------------------------------
// SegmentVQVAE forward on MI455X (gfx1250, wave32, WMMA).
// bf16 WMMA implicit-GEMM, double-buffered LDS tiles, grouped fragment preload
// pinned with sched barriers so WMMAs issue back-to-back.
// ---------------------------------------------------------------------------

#define BB   128   // batch
#define TT   256   // input seq len
#define DD   1024  // input dim
#define LL   128   // model dim
#define NH   4     // heads
#define DH   32    // head dim
#define T1   128   // after conv1 (stride 2, pad 2)
#define T2   64    // after conv2

typedef __attribute__((ext_vector_type(16))) __bf16        v16bf;
typedef __attribute__((ext_vector_type(8)))  float         v8f;
typedef __attribute__((ext_vector_type(4)))  float         f32x4;
typedef __attribute__((ext_vector_type(4)))  unsigned int  u32x4;

#if defined(__gfx1250__) && __has_builtin(__builtin_amdgcn_global_load_async_to_lds_b128)
#define HAVE_ASYNC_LDS 1
// exact parameter types per hipcc diagnostic: int __vector(4) in AS1 / AS3
typedef int vint4 __attribute__((vector_size(16)));
typedef __attribute__((address_space(1))) vint4* gas1_i4;
typedef __attribute__((address_space(3))) vint4* las3_i4;
#endif

#if __has_builtin(__builtin_amdgcn_sched_barrier)
#define SCHED_FENCE() __builtin_amdgcn_sched_barrier(0)
#else
#define SCHED_FENCE()
#endif

__device__ __forceinline__ unsigned pack2bf(float a, float b) {
#if defined(__gfx1250__) && __has_builtin(__builtin_amdgcn_cvt_pk_bf16_f32)
  auto r = __builtin_amdgcn_cvt_pk_bf16_f32(a, b);   // v_cvt_pk_bf16_f32
  unsigned u;
  __builtin_memcpy(&u, &r, 4);
  return u;
#else
  unsigned ua = __float_as_uint(a);
  ua += 0x7FFFu + ((ua >> 16) & 1u);                 // RNE
  unsigned ub = __float_as_uint(b);
  ub += 0x7FFFu + ((ub >> 16) & 1u);
  return (ua >> 16) | (ub & 0xFFFF0000u);
#endif
}

__device__ __forceinline__ float gelu_exact(float x) {
  return 0.5f * x * (1.0f + erff(x * 0.70710678118654752f));
}

union FragAB { v16bf v; u32x4 u[2]; };

// ---------------------------------------------------------------------------
// k_gemm_t<MODE>: C[M,N] = epilogue(A_im2col[M,K] @ W[N,K]^T + bias)
//   MODE 0: dense A (row stride lda, element offset aoff)
//   MODE 1: conv1 im2col over X0/X1/X2   (D=1024, k=5, stride 2, pad 2)
//   MODE 2: conv2 im2col over A (=H1)    (C=128,  k=5, stride 2, pad 2)
//   act: 0 none, 1 relu, 2 gelu ; addpe: + pos_scale * sinusoidal PE (t=row%64)
// Tiling: 128x128 block tile, K-step 32, 8 waves x (16 rows x 128 cols).
// Pipeline per K-step: regs->LDS(p) | barrier | issue next global loads |
// {load 4 B frags | fence | 4 WMMAs} x2  (2 dscnt waits per K-step).
// ---------------------------------------------------------------------------
template <int MODE>
__global__ __launch_bounds__(256, 1)
void k_gemm_t(const float* __restrict__ A, long lda, long aoff,
              const float* __restrict__ X0, const float* __restrict__ X1,
              const float* __restrict__ X2,
              const float* __restrict__ W, const float* __restrict__ bias,
              const float* __restrict__ psp,
              float* __restrict__ C, int M, int N, int K,
              int act, int addpe) {
  __shared__ unsigned short sA[2][128][40];   // 2 x (128 rows x 32 K, pad 8) bf16
  __shared__ unsigned short sB[2][128][40];   // 2 x (128 cols x 32 K) bf16

  const int tid   = threadIdx.x;
  const int m0    = blockIdx.x * 128;
  const int n0    = blockIdx.y * 128;
  const int arow  = tid >> 1;              // 0..127
  const int acol0 = (tid & 1) * 16;        // 0 or 16

  const int wv = tid >> 5;                 // wave 0..7 -> 16-row strip
  const int ln = tid & 31;
  const int lr = ln & 15;
  const int lh = ln >> 4;

  v8f acc[8];
#pragma unroll
  for (int j = 0; j < 8; ++j)
#pragma unroll
    for (int e = 0; e < 8; ++e) acc[j][e] = 0.0f;

  float ra[16], rw[16];                    // staging registers (next tile)

  auto load_tiles = [&](int kb) {
    // W tile: rows n0+arow, K-contiguous (exactly the WMMA B layout)
    const float* wp = W + (size_t)(n0 + arow) * (size_t)K + kb + acol0;
#pragma unroll
    for (int v = 0; v < 4; ++v) {
      f32x4 f = *(const f32x4*)(wp + v * 4);
      rw[v * 4 + 0] = f.x; rw[v * 4 + 1] = f.y;
      rw[v * 4 + 2] = f.z; rw[v * 4 + 3] = f.w;
    }
    if (kb + 32 < K) __builtin_prefetch(wp + 32, 0, 1);   // global_prefetch_b8
    // A tile
    if (MODE == 0) {
      const float* ap = A + aoff + (size_t)(m0 + arow) * (size_t)lda + kb + acol0;
#pragma unroll
      for (int v = 0; v < 4; ++v) {
        f32x4 f = *(const f32x4*)(ap + v * 4);
        ra[v * 4 + 0] = f.x; ra[v * 4 + 1] = f.y;
        ra[v * 4 + 2] = f.z; ra[v * 4 + 3] = f.w;
      }
    } else if (MODE == 1) {                 // conv1 im2col: k = c*5+kk
      const int m   = m0 + arow;
      const int s   = m >> 14;              // / (B*T1)
      const int rem = m & 16383;
      const int b   = rem >> 7;
      const int t   = rem & 127;
      const float* xs = (s == 0) ? X0 : ((s == 1) ? X1 : X2);
      int c  = (kb + acol0) / 5;
      int kk = (kb + acol0) - 5 * c;
#pragma unroll
      for (int i = 0; i < 16; ++i) {
        const int tin = 2 * t + kk - 2;
        ra[i] = (tin >= 0 && tin < TT)
                    ? xs[((size_t)b * TT + tin) * DD + c] : 0.0f;
        if (++kk == 5) { kk = 0; ++c; }
      }
    } else {                                // conv2 im2col over H1 (3B,128,128)
      const int m  = m0 + arow;
      const int sb = m >> 6;
      const int t  = m & 63;
      int c  = (kb + acol0) / 5;
      int kk = (kb + acol0) - 5 * c;
#pragma unroll
      for (int i = 0; i < 16; ++i) {
        const int tin = 2 * t + kk - 2;
        ra[i] = (tin >= 0 && tin < T1)
                    ? A[((size_t)sb * T1 + tin) * LL + c] : 0.0f;
        if (++kk == 5) { kk = 0; ++c; }
      }
    }
  };

  auto store_tiles = [&](int p) {
#pragma unroll
    for (int i = 0; i < 16; i += 2) {
      *(unsigned*)&sA[p][arow][acol0 + i] = pack2bf(ra[i], ra[i + 1]);
      *(unsigned*)&sB[p][arow][acol0 + i] = pack2bf(rw[i], rw[i + 1]);
    }
  };

  auto compute = [&](int p) {
    FragAB af;
    af.u[0] = *(const u32x4*)&sA[p][wv * 16 + lr][lh * 8];
    af.u[1] = *(const u32x4*)&sA[p][wv * 16 + lr][lh * 8 + 16];
#pragma unroll
    for (int g = 0; g < 2; ++g) {           // two groups of 4 B fragments
      FragAB bf[4];
#pragma unroll
      for (int j = 0; j < 4; ++j) {
        const int jj = g * 4 + j;
        bf[j].u[0] = *(const u32x4*)&sB[p][jj * 16 + lr][lh * 8];
        bf[j].u[1] = *(const u32x4*)&sB[p][jj * 16 + lr][lh * 8 + 16];
      }
      SCHED_FENCE();                        // pin loads above the WMMA chain
#pragma unroll
      for (int j = 0; j < 4; ++j)           // 4 back-to-back WMMAs
        acc[g * 4 + j] = __builtin_amdgcn_wmma_f32_16x16x32_bf16(
            false, af.v, false, bf[j].v, (short)0, acc[g * 4 + j], false, false);
    }
  };

  load_tiles(0);
  int p = 0;
  for (int kb = 0; kb < K; kb += 32) {
    store_tiles(p);
    __syncthreads();                        // one barrier per K-step
    if (kb + 32 < K) load_tiles(kb + 32);   // global loads fly during compute
    compute(p);
    p ^= 1;
  }

  // ---- epilogue: bias + activation (+ sinusoidal PE), fp32 store ----
  const float ps = (addpe && psp) ? psp[0] : 0.0f;
#pragma unroll
  for (int j = 0; j < 8; ++j) {
    const int col = n0 + j * 16 + lr;
    const float bv = bias ? bias[col] : 0.0f;
#pragma unroll
    for (int r = 0; r < 8; ++r) {
      const int row = m0 + wv * 16 + lh * 8 + r;
      float x = acc[j][r] + bv;
      if (act == 1)      x = fmaxf(x, 0.0f);
      else if (act == 2) x = gelu_exact(x);
      if (addpe) {
        const int t   = row & (T2 - 1);
        const float fi = (float)(col & ~1);
        const float dv = __expf(-9.210340372f * fi * (1.0f / (float)LL));
        const float ang = (float)t * dv;
        x += ps * ((col & 1) ? __cosf(ang) : __sinf(ang));
      }
      C[(size_t)row * N + col] = x;
    }
  }
}

// ---------------------------------------------------------------------------
// Attention: per (batch,head) block; S<=64, dh=32. QKV rows: [tok][3*128].
// ---------------------------------------------------------------------------
__global__ __launch_bounds__(64)
void k_attn(const float* __restrict__ qkv, float* __restrict__ out, int S) {
  const int b = blockIdx.x / NH;
  const int h = blockIdx.x % NH;
  __shared__ float kb[64 * DH];
  __shared__ float vb[64 * DH];
  __shared__ float sc[64 * 64];

  for (int idx = threadIdx.x; idx < S * DH; idx += 64) {
    const int t = idx >> 5, d = idx & 31;
    const size_t base = (size_t)(b * S + t) * (3 * LL) + h * DH + d;
    kb[idx] = qkv[base + LL];
    vb[idx] = qkv[base + 2 * LL];
  }
  __syncthreads();

  const int t = threadIdx.x;
  if (t < S) {
    float q[DH];
#pragma unroll
    for (int d = 0; d < DH; ++d)
      q[d] = qkv[(size_t)(b * S + t) * (3 * LL) + h * DH + d];

    float mx = -1e30f;
    for (int u = 0; u < S; ++u) {
      float s = 0.0f;
#pragma unroll
      for (int d = 0; d < DH; ++d) s += q[d] * kb[u * DH + d];
      s *= 0.1767766953f;                   // 1/sqrt(32)
      sc[t * 64 + u] = s;
      mx = fmaxf(mx, s);
    }
    float den = 0.0f;
    for (int u = 0; u < S; ++u) {
      float e = __expf(sc[t * 64 + u] - mx);
      sc[t * 64 + u] = e;
      den += e;
    }
    const float inv = 1.0f / den;
    float o[DH];
#pragma unroll
    for (int d = 0; d < DH; ++d) o[d] = 0.0f;
    for (int u = 0; u < S; ++u) {
      const float w = sc[t * 64 + u] * inv;
#pragma unroll
      for (int d = 0; d < DH; ++d) o[d] += w * vb[u * DH + d];
    }
#pragma unroll
    for (int d = 0; d < DH; ++d)
      out[(size_t)(b * S + t) * LL + h * DH + d] = o[d];
  }
}

// residual add + layernorm over rows of 128
__global__ __launch_bounds__(128)
void k_add_ln(const float* __restrict__ X, const float* __restrict__ Y,
              const float* __restrict__ w, const float* __restrict__ b,
              float* __restrict__ out) {
  const int row = blockIdx.x, c = threadIdx.x;
  const size_t idx = (size_t)row * LL + c;
  const float v = X[idx] + Y[idx];
  __shared__ float s1[128], s2[128];
  s1[c] = v; s2[c] = v * v;
  __syncthreads();
  for (int st = 64; st > 0; st >>= 1) {
    if (c < st) { s1[c] += s1[c + st]; s2[c] += s2[c + st]; }
    __syncthreads();
  }
  const float m   = s1[0] * (1.0f / LL);
  const float var = s2[0] * (1.0f / LL) - m * m;
  out[idx] = (v - m) * rsqrtf(var + 1e-5f) * w[c] + b[c];
}

__global__ __launch_bounds__(128)
void k_pool(const float* __restrict__ X, float* __restrict__ P) {
  const int sb = blockIdx.x, c = threadIdx.x;
  float s = 0.0f;
  for (int t = 0; t < T2; ++t) s += X[((size_t)sb * T2 + t) * LL + c];
  P[(size_t)sb * LL + c] = s * (1.0f / T2);
}

// decoder input: Z[(b*3+j)] = FC[(j*B+b)] + ps * PE_DEC[j]
__global__ __launch_bounds__(128)
void k_zpe(const float* __restrict__ FC, const float* __restrict__ psp,
           float* __restrict__ Z) {
  const int bj = blockIdx.x;
  const int b = bj / 3, j = bj % 3, c = threadIdx.x;
  const float fi = (float)(c & ~1);
  const float dv = __expf(-9.210340372f * fi * (1.0f / (float)LL));
  const float ang = (float)j * dv;
  const float pe = (c & 1) ? __cosf(ang) : __sinf(ang);
  Z[(size_t)bj * LL + c] = FC[((size_t)j * BB + b) * LL + c] + psp[0] * pe;
}

// conv_transpose1d (k=4, stride=2, pad=1) + gelu : x[b,c,Lin] -> y[b,n,Lout]
__global__ void k_ct(const float* __restrict__ x, const float* __restrict__ w,
                     const float* __restrict__ bias, float* __restrict__ y,
                     int Lin, int Lout) {
  const int b = blockIdx.x;
  __shared__ float xl[256];
  for (int i = threadIdx.x; i < LL * Lin; i += blockDim.x)
    xl[i] = x[(size_t)b * LL * Lin + i];
  __syncthreads();
  const int n = threadIdx.x / Lout;
  const int t = threadIdx.x % Lout;
  float s = bias[n];
  for (int p = 0; p < Lin; ++p) {
    const int wi = t + 1 - 2 * p;            // flipped-kernel index
    if (wi >= 0 && wi < 4) {
      for (int c = 0; c < LL; ++c)
        s += xl[c * Lin + p] * w[((size_t)c * LL + n) * 4 + wi];
    }
  }
  y[((size_t)b * LL + n) * Lout + t] = gelu_exact(s);
}

// conv1d k=3 pad=1 (128 -> 1024 channels, len 4), output [b][t][n]
__global__ __launch_bounds__(256)
void k_conv3(const float* __restrict__ x, const float* __restrict__ w,
             const float* __restrict__ bias, float* __restrict__ rc) {
  const int b = blockIdx.x >> 2;
  const int t = blockIdx.x & 3;
  __shared__ float xl[512];
  for (int i = threadIdx.x; i < 512; i += 256) xl[i] = x[(size_t)b * 512 + i];
  __syncthreads();
  for (int n = threadIdx.x; n < DD; n += 256) {
    float s = bias[n];
    for (int kk = 0; kk < 3; ++kk) {
      const int ti = t - 1 + kk;
      if (ti >= 0 && ti < 4) {
        for (int c = 0; c < LL; ++c)
          s += xl[c * 4 + ti] * w[((size_t)n * LL + c) * 3 + kk];
      }
    }
    rc[((size_t)b * 4 + t) * DD + n] = s;
  }
}

__global__ void k_zero(float* out) { out[0] = 0.0f; }

// L1 loss vs tokens_curr; recon is non-zero only for t<4. The big stream of
// `curr` goes through the gfx1250 async global->LDS path when available.
__global__ __launch_bounds__(256)
void k_loss(const float* __restrict__ curr, const float* __restrict__ rc,
            float* __restrict__ out) {
  const int bt = blockIdx.x;                 // b*256 + t
  const int b = bt >> 8, t = bt & 255;
  float s = 0.0f;
#ifdef HAVE_ASYNC_LDS
  __shared__ float buf[1024];
  {
    float* gp = const_cast<float*>(curr) + (size_t)bt * DD + threadIdx.x * 4;
    gas1_i4 g = (gas1_i4)gp;                       // addrspace(1) int4*
    las3_i4 l = (las3_i4)&buf[threadIdx.x * 4];    // addrspace(3) int4*
    __builtin_amdgcn_global_load_async_to_lds_b128(g, l, 0, 0);
    asm volatile("s_wait_asynccnt 0x0" ::: "memory");
    __syncthreads();
  }
  for (int n = threadIdx.x; n < DD; n += 256) {
    const float c = buf[n];
    const float r = (t < 4) ? rc[((size_t)b * 4 + t) * DD + n] : 0.0f;
    s += fabsf(r - c);
  }
#else
  for (int n = threadIdx.x; n < DD; n += 256) {
    const float c = curr[(size_t)bt * DD + n];
    const float r = (t < 4) ? rc[((size_t)b * 4 + t) * DD + n] : 0.0f;
    s += fabsf(r - c);
  }
#endif
  __shared__ float red[256];
  red[threadIdx.x] = s;
  __syncthreads();
  for (int st = 128; st > 0; st >>= 1) {
    if (threadIdx.x < st) red[threadIdx.x] += red[threadIdx.x + st];
    __syncthreads();
  }
  if (threadIdx.x == 0) atomicAdd(out, red[0]);
}

__global__ void k_final(float* out) {
  out[0] *= (1.0f / ((float)BB * (float)TT * (float)DD));
}

// ---------------------------------------------------------------------------
extern "C" void kernel_launch(void* const* d_in, const int* in_sizes, int n_in,
                              void* d_out, int out_size, void* d_ws, size_t ws_size,
                              hipStream_t stream) {
  (void)in_sizes; (void)n_in; (void)out_size; (void)ws_size;
  auto IN = [&](int i) { return (const float*)d_in[i]; };

  const float* Xp = IN(0);
  const float* Xc = IN(1);
  const float* Xn = IN(2);
  // enc_params: 3..33, dec_params: 34..66 (flattened in dict/list order)
  const int ENC_L0 = 8, DEC_L0 = 35;        // layer dict bases (12 entries each)

  float* ws = (float*)d_ws;
  float* H1   = ws;                         // 3B*128*128 = 6,291,456
  float* Xb   = H1  + 6291456;              // 24576*128  = 3,145,728
  float* QKV  = Xb  + 3145728;              // 24576*384  = 9,437,184
  float* ATT  = QKV + 9437184;              // 24576*128
  float* TMP  = ATT + 3145728;              // 24576*128
  float* FF1  = TMP + 3145728;              // 24576*256  = 6,291,456
  float* PO   = FF1 + 6291456;              // pooled 384*128
  float* FC   = PO  + 49152;                // fc out 384*128
  float* Z    = FC  + 49152;                // decoder tokens 384*128
  float* PR   = Z   + 49152;                // proj out 128*128
  float* Y1   = PR  + 16384;                // 128*128*2
  float* Y2   = Y1  + 32768;                // 128*128*4
  float* RC   = Y2  + 65536;                // recon 128*4*1024

  float* OUT = (float*)d_out;

  auto gemm = [&](const float* A, long lda, long aoff, const float* W,
                  const float* bias, const float* psp, float* C,
                  int M, int N, int K, int mode, int act, int addpe) {
    dim3 g(M / 128, N / 128);
    if (mode == 0)
      k_gemm_t<0><<<g, 256, 0, stream>>>(A, lda, aoff, Xp, Xc, Xn, W, bias, psp,
                                         C, M, N, K, act, addpe);
    else if (mode == 1)
      k_gemm_t<1><<<g, 256, 0, stream>>>(A, lda, aoff, Xp, Xc, Xn, W, bias, psp,
                                         C, M, N, K, act, addpe);
    else
      k_gemm_t<2><<<g, 256, 0, stream>>>(A, lda, aoff, Xp, Xc, Xn, W, bias, psp,
                                         C, M, N, K, act, addpe);
  };

  const int Me = 3 * BB * T2;               // 24576 transformer tokens
  const int Md = BB * 3;                    // 384 decoder tokens

  // ---- encoder convs (all 3 streams fused in M) ----
  gemm(nullptr, 0, 0, IN(3), IN(4), nullptr, H1, 3 * BB * T1, LL, DD * 5, 1, 2, 0);
  gemm(H1, 0, 0, IN(5), IN(6), IN(7), Xb, Me, LL, LL * 5, 2, 2, 1);

  // ---- encoder transformer layers ----
  for (int l = 0; l < 2; ++l) {
    const int p = ENC_L0 + 12 * l;
    gemm(Xb, LL, 0, IN(p + 0), IN(p + 1), nullptr, QKV, Me, 3 * LL, LL, 0, 0, 0);
    k_attn<<<dim3(3 * BB * NH), 64, 0, stream>>>(QKV, ATT, T2);
    gemm(ATT, LL, 0, IN(p + 2), IN(p + 3), nullptr, TMP, Me, LL, LL, 0, 0, 0);
    k_add_ln<<<dim3(Me), 128, 0, stream>>>(Xb, TMP, IN(p + 4), IN(p + 5), Xb);
    gemm(Xb, LL, 0, IN(p + 6), IN(p + 7), nullptr, FF1, Me, 2 * LL, LL, 0, 1, 0);
    gemm(FF1, 2 * LL, 0, IN(p + 8), IN(p + 9), nullptr, TMP, Me, LL, 2 * LL, 0, 0, 0);
    k_add_ln<<<dim3(Me), 128, 0, stream>>>(Xb, TMP, IN(p + 10), IN(p + 11), Xb);
  }

  // ---- pool + fc -> latents ----
  k_pool<<<dim3(3 * BB), 128, 0, stream>>>(Xb, PO);
  gemm(PO, LL, 0, IN(32), IN(33), nullptr, FC, 3 * BB, LL, LL, 0, 0, 0);

  // ---- decoder tokens (stack + PE) ----
  k_zpe<<<dim3(Md), 128, 0, stream>>>(FC, IN(34), Z);

  // ---- decoder transformer layers ----
  for (int l = 0; l < 2; ++l) {
    const int p = DEC_L0 + 12 * l;
    gemm(Z, LL, 0, IN(p + 0), IN(p + 1), nullptr, QKV, Md, 3 * LL, LL, 0, 0, 0);
    k_attn<<<dim3(BB * NH), 64, 0, stream>>>(QKV, ATT, 3);
    gemm(ATT, LL, 0, IN(p + 2), IN(p + 3), nullptr, TMP, Md, LL, LL, 0, 0, 0);
    k_add_ln<<<dim3(Md), 128, 0, stream>>>(Z, TMP, IN(p + 4), IN(p + 5), Z);
    gemm(Z, LL, 0, IN(p + 6), IN(p + 7), nullptr, FF1, Md, 2 * LL, LL, 0, 1, 0);
    gemm(FF1, 2 * LL, 0, IN(p + 8), IN(p + 9), nullptr, TMP, Md, LL, 2 * LL, 0, 0, 0);
    k_add_ln<<<dim3(Md), 128, 0, stream>>>(Z, TMP, IN(p + 10), IN(p + 11), Z);
  }

  // ---- proj of center token (strided A: row b = Z[b*3+1]) ----
  gemm(Z, 3 * LL, LL, IN(59), IN(60), nullptr, PR, BB, LL, LL, 0, 0, 0);

  // ---- transposed convs + conv3 ----
  k_ct<<<dim3(BB), 256, 0, stream>>>(PR, IN(61), IN(62), Y1, 1, 2);
  k_ct<<<dim3(BB), 512, 0, stream>>>(Y1, IN(63), IN(64), Y2, 2, 4);
  k_conv3<<<dim3(BB * 4), 256, 0, stream>>>(Y2, IN(65), IN(66), RC);

  // ---- L1 loss over full (B,T,D); recon is zero for t>=4 ----
  k_zero<<<1, 1, 0, stream>>>(OUT);
  k_loss<<<dim3(BB * TT), 256, 0, stream>>>(Xc, RC, OUT);
  k_final<<<1, 1, 0, stream>>>(OUT);
}
